// BiLSTMCRF_22428319219956
// MI455X (gfx1250) — matrix operand order repaired and checked by reading, not compiled, not run on hardware
//
#include <hip/hip_runtime.h>
#include <hip/hip_bf16.h>
#include <math.h>

typedef __bf16 v16bf __attribute__((ext_vector_type(16)));
typedef float  v8f   __attribute__((ext_vector_type(8)));

#define DEVINL static __device__ __forceinline__

// ---------------- problem dims ----------------
constexpr int EMB   = 256;
constexpr int HIDC  = 512;          // concat hidden (2 dirs)
constexpr int NL    = 9;            // labels
constexpr int NB    = 64;           // batch
constexpr int NT    = 512;          // seq len
constexpr int NALL  = 2048;         // 4H per dir * 2 dirs
constexpr int MROWS = NB * NT;      // 32768 flat rows

// ---------------- workspace layout (bytes) ----------------
constexpr size_t WT_SW_OFF  = 0;                                   // 1 MB   bf16 W_ih (both dirs), B-swizzled tiles
constexpr size_t WHH_SW_OFF = 1ull << 20;                          // 1 MB   bf16 W_hh (both dirs), B-swizzled tiles
constexpr size_t CLS_SW_OFF = 2ull << 20;                          // 16 KB  bf16 W_cls padded N=16, B-swizzled
constexpr size_t BIAS_OFF   = CLS_SW_OFF + 16384;                  // 8 KB   f32 (b_ih + b_hh) both dirs
constexpr size_t SYNC_OFF   = BIAS_OFF + 8192;                     // 256 B  spin-barrier counters
constexpr size_t HX_OFF     = SYNC_OFF + 256;                      // 128 KB h exchange, double buffered
// xg stored as WMMA C/D-layout tiles: [t][n-tile(128)][b-tile(4)][lane(32)][reg(8)] bf16
constexpr size_t XG_OFF     = 4ull << 20;                          // 128 MB
constexpr size_t HALL_OFF   = XG_OFF + (size_t)MROWS * NALL * 2;   // 32 MB  bf16 h[m][512] row-major
constexpr size_t EMIS_OFF   = HALL_OFF + (size_t)MROWS * HIDC * 2; // 2 MB   f32 emissions[m][16]
// total ~166 MB of d_ws assumed available

// ---------------- helpers ----------------
DEVINL v8f wmma_bf16(v16bf a, v16bf b, v8f c) {
  return __builtin_amdgcn_wmma_f32_16x16x32_bf16(false, a, false, b, (short)0, c, false, false);
}
DEVINL float sigmoidf(float x) { return 1.f / (1.f + __expf(-x)); }

// A operand (16x32 bf16) from a row-contiguous bf16 row: two 16B runs per lane.
DEVINL v16bf load_a_bf16_row(const __bf16* rowPtr, int kt, int hi) {
  v16bf a;
  ((uint4*)&a)[0] = *(const uint4*)(rowPtr + kt * 32 + hi * 8);
  ((uint4*)&a)[1] = *(const uint4*)(rowPtr + kt * 32 + hi * 8 + 16);
  return a;
}
// A operand from an f32 row (embedding gather), converting to bf16 on the fly.
DEVINL v16bf load_a_f32_row(const float* rowPtr, int kt, int hi) {
  const float4* p0 = (const float4*)(rowPtr + kt * 32 + hi * 8);
  const float4* p1 = (const float4*)(rowPtr + kt * 32 + hi * 8 + 16);
  float4 f0 = p0[0], f1 = p0[1], f2 = p1[0], f3 = p1[1];
  v16bf a;
  a[0]=(__bf16)f0.x; a[1]=(__bf16)f0.y; a[2]=(__bf16)f0.z; a[3]=(__bf16)f0.w;
  a[4]=(__bf16)f1.x; a[5]=(__bf16)f1.y; a[6]=(__bf16)f1.z; a[7]=(__bf16)f1.w;
  a[8]=(__bf16)f2.x; a[9]=(__bf16)f2.y; a[10]=(__bf16)f2.z; a[11]=(__bf16)f2.w;
  a[12]=(__bf16)f3.x; a[13]=(__bf16)f3.y; a[14]=(__bf16)f3.z; a[15]=(__bf16)f3.w;
  return a;
}
// byte offset (in elements) of a D-layout xg tile
DEVINL size_t xg_tile(int t, int nt, int bt) {
  return (((size_t)t * 128 + nt) * 4 + bt) * 256;
}

// ---------------- kernel 0: weight conversion + B-swizzle ----------------
__global__ __launch_bounds__(256) void k_prep(
    const float* __restrict__ wihf, const float* __restrict__ whhf,
    const float* __restrict__ bihf, const float* __restrict__ bhhf,
    const float* __restrict__ wihb, const float* __restrict__ whhb,
    const float* __restrict__ bihb, const float* __restrict__ bhhb,
    const float* __restrict__ wcls, unsigned char* ws) {
  const int N_WT = 2048 * 256, N_WHH = 2 * 1024 * 256, N_CLS = 16 * 512, N_BIAS = 2048;
  int idx = blockIdx.x * 256 + threadIdx.x;
  if (idx < N_WT) {                       // W_ih^T as B tiles (K=256, N=2048)
    int n = idx / 256, k = idx % 256;
    int dir = n >> 10, r = n & 1023;
    float v = dir ? wihb[r * 256 + k] : wihf[r * 256 + k];
    int nt = n >> 4, kt = k >> 5, kk = k & 31;
    int lane = (n & 15) | (((kk >> 3) & 1) << 4), elem = (kk & 7) | ((kk >> 4) << 3);
    ((__bf16*)(ws + WT_SW_OFF))[(size_t)(nt * 8 + kt) * 512 + lane * 16 + elem] = (__bf16)v;
    return;
  }
  idx -= N_WT;
  if (idx < N_WHH) {                      // W_hh^T as B tiles per (dir, j-quarter)
    int dir = idx / (1024 * 256); int rem = idx % (1024 * 256);
    int n = rem / 256, k = rem % 256;
    int g = n >> 8, j = n & 255;
    int q = j >> 6, jl = j & 63, jt = jl >> 4, nn = jl & 15;
    int lt = g * 4 + jt, kt = k >> 5, kk = k & 31;
    int lane = nn | (((kk >> 3) & 1) << 4), elem = (kk & 7) | ((kk >> 4) << 3);
    float v = dir ? whhb[n * 256 + k] : whhf[n * 256 + k];
    int blk = (dir * 4 + q) * 128 + lt * 8 + kt;
    ((__bf16*)(ws + WHH_SW_OFF))[(size_t)blk * 512 + lane * 16 + elem] = (__bf16)v;
    return;
  }
  idx -= N_WHH;
  if (idx < N_CLS) {                      // W_cls^T padded to N=16 as B tiles (K=512)
    int n = idx / 512, k = idx % 512;
    float v = (n < NL) ? wcls[n * 512 + k] : 0.f;
    int kt = k >> 5, kk = k & 31;
    int lane = n | (((kk >> 3) & 1) << 4), elem = (kk & 7) | ((kk >> 4) << 3);
    ((__bf16*)(ws + CLS_SW_OFF))[(size_t)kt * 512 + lane * 16 + elem] = (__bf16)v;
    return;
  }
  idx -= N_CLS;
  if (idx < N_BIAS) {                     // combined gate bias
    int dir = idx >> 10, r = idx & 1023;
    ((float*)(ws + BIAS_OFF))[idx] = dir ? (bihb[r] + bhhb[r]) : (bihf[r] + bhhf[r]);
  }
}

// ---------------- kernel 1: xg = gather(emb) @ [W_ih_f | W_ih_b]^T + bias ----------------
// One workgroup per timestep t (blockIdx.x) x n-block (blockIdx.y). M-dim is the
// batch at fixed t, so the output tiles are exactly the C-operand tiles the
// recurrence kernel consumes; they are stored in D-register layout so each lane
// stores/loads one contiguous 16B run per tile.
__global__ __launch_bounds__(256) void k_xg(const int* __restrict__ ids,
                                            const float* __restrict__ emb,
                                            unsigned char* ws) {
  __shared__ __bf16 sB[8 * 8 * 512];      // 64 KB: 8 n-tiles x 8 k-tiles, B-swizzled
  int tid = threadIdx.x, lane = tid & 31, wv = tid >> 5;
  int bt = wv & 3, ng = wv >> 2, lnt0 = ng * 4;
  int t = blockIdx.x;
  {
    const uint4* src = (const uint4*)(ws + WT_SW_OFF + (size_t)blockIdx.y * 65536);
    uint4* dst = (uint4*)sB;
    for (int i = tid; i < 4096; i += 256) dst[i] = src[i];
  }
  __syncthreads();
  int hi = lane >> 4;
  int b = bt * 16 + (lane & 15);
  const float* erow = emb + (size_t)ids[b * NT + t] * EMB;   // fused embedding gather
  v8f acc[4];
  for (int qq = 0; qq < 4; ++qq) for (int e = 0; e < 8; ++e) acc[qq][e] = 0.f;
  for (int kt = 0; kt < 8; ++kt) {
    v16bf a = load_a_f32_row(erow, kt, hi);
    for (int qq = 0; qq < 4; ++qq) {
      v16bf bm = *(const v16bf*)(sB + (size_t)((lnt0 + qq) * 8 + kt) * 512 + lane * 16);
      acc[qq] = wmma_bf16(a, bm, acc[qq]);
    }
  }
  const float* biasAll = (const float*)(ws + BIAS_OFF);
  __bf16* xg = (__bf16*)(ws + XG_OFF);
  for (int qq = 0; qq < 4; ++qq) {
    int nt = blockIdx.y * 8 + (lnt0 + qq);
    float bias = biasAll[nt * 16 + (lane & 15)];
    __bf16 pack[8];
    for (int r = 0; r < 8; ++r) pack[r] = (__bf16)(acc[qq][r] + bias);
    *(uint4*)(xg + xg_tile(t, nt, bt) + lane * 8) = *(const uint4*)pack;  // one b128 store
  }
}

// ---------------- kernel 2: persistent BiLSTM recurrence ----------------
// 16 WGs: dir(2) x j-quarter(4) x batch-half(2). W_hh slice resident in LDS,
// cell state in VGPRs, h exchanged across quarter-WGs each step via global
// double buffer + release/acquire spin barrier. C operand comes straight from
// the D-layout xg tiles: one contiguous b128 load per gate per lane.
__global__ __launch_bounds__(256) void k_lstm(unsigned char* ws) {
  __shared__ __bf16 sW[128 * 512];   // 128 KB B-swizzled W_hh slice
  __shared__ __bf16 sHA[16 * 512];   // 16 KB  A-swizzled h (32 batch rows x K=256)
  int bid = blockIdx.x;
  int dir = bid & 1, q = (bid >> 1) & 3, bg = (bid >> 3) & 1;
  int tid = threadIdx.x, lane = tid & 31, wv = tid >> 5;
  int mt = wv & 1, jt = wv >> 1;           // wave owns (m-tile, j-tile) and all 4 gates
  int hi = lane >> 4, nn = lane & 15;
  {
    const uint4* src = (const uint4*)(ws + WHH_SW_OFF + (size_t)(dir * 4 + q) * 131072);
    uint4* dst = (uint4*)sW;
    for (int i = tid; i < 8192; i += 256) dst[i] = src[i];
  }
  __syncthreads();
  __bf16* xg   = (__bf16*)(ws + XG_OFF);
  __bf16* hall = (__bf16*)(ws + HALL_OFF);
  __bf16* hx   = (__bf16*)(ws + HX_OFF);
  unsigned int* cnt = (unsigned int*)(ws + SYNC_OFF) + (dir * 2 + bg);
  int bt = bg * 2 + mt;                    // batch-tile index in xg layout
  int ntg[4];
  for (int g = 0; g < 4; ++g) ntg[g] = dir * 64 + g * 16 + q * 4 + jt;
  int jglob = q * 64 + jt * 16 + nn;
  int kt_h = jglob >> 5, kkh = jglob & 31;
  int elem_h = (kkh & 7) | (((kkh >> 4) & 1) << 3);
  int laneh_hi = ((kkh >> 3) & 1) << 4;
  float c_reg[8];
  for (int r = 0; r < 8; ++r) c_reg[r] = 0.f;

  for (int step = 0; step < NT; ++step) {
    int t = dir ? (NT - 1 - step) : step;
    // C operand := xg_t (bias folded in); contiguous 16B per gate per lane
    v8f acc[4];
    for (int g = 0; g < 4; ++g) {
      uint4 v = *(const uint4*)(xg + xg_tile(t, ntg[g], bt) + lane * 8);
      const __bf16* pv = (const __bf16*)&v;
      for (int r = 0; r < 8; ++r) acc[g][r] = (float)pv[r];
    }
    if (step + 1 < NT) {                 // prefetch next step's xg (global_prefetch_b8)
      int tn = dir ? (t - 1) : (t + 1);
      for (int g = 0; g < 4; ++g)
        __builtin_prefetch(xg + xg_tile(tn, ntg[g], bt) + lane * 8, 0, 1);
    }
    if (step > 0) {                      // += h_{t-1} @ W_hh^T
      for (int kt = 0; kt < 8; ++kt) {
        v16bf a = *(const v16bf*)(sHA + (size_t)(mt * 8 + kt) * 512 + lane * 16);
        for (int g = 0; g < 4; ++g) {
          v16bf bm = *(const v16bf*)(sW + (size_t)((g * 4 + jt) * 8 + kt) * 512 + lane * 16);
          acc[g] = wmma_bf16(a, bm, acc[g]);
        }
      }
    }
    __syncthreads();                     // all waves done reading sHA
    int par = step & 1;
    __bf16* hxo = hx + (size_t)(((par * 2 + dir) * 2 + bg) * 4 + q) * 2048;
    for (int r = 0; r < 8; ++r) {        // gates: i,f,g,o  ->  c,h
      float ig = sigmoidf(acc[0][r]);
      float fg = sigmoidf(acc[1][r]);
      float gg = tanhf(acc[2][r]);
      float og = sigmoidf(acc[3][r]);
      float c  = fg * c_reg[r] + ig * gg;
      c_reg[r] = c;
      __bf16 hb = (__bf16)(og * tanhf(c));
      int m_local = hi * 8 + r;          // 0..15 within m-tile
      int lane_a  = m_local | laneh_hi;
      sHA[(size_t)(mt * 8 + kt_h) * 512 + lane_a * 16 + elem_h] = hb;   // own h into LDS (A layout)
      hxo[(mt * 16 + m_local) * 64 + (jglob & 63)] = hb;                // export for partner WGs
      int b = bg * 32 + mt * 16 + m_local;
      hall[((size_t)b * NT + t) * HIDC + dir * 256 + jglob] = hb;       // final h output
    }
    __threadfence();
    __syncthreads();
    if (tid == 0) {                      // pair-group spin barrier (4 WGs per group)
      __hip_atomic_fetch_add(cnt, 1u, __ATOMIC_RELEASE, __HIP_MEMORY_SCOPE_AGENT);
      unsigned int target = 4u * (unsigned int)(step + 1);
      while (__hip_atomic_load(cnt, __ATOMIC_ACQUIRE, __HIP_MEMORY_SCOPE_AGENT) < target)
        __builtin_amdgcn_s_sleep(1);
    }
    __syncthreads();
    // import the other 3 quarters' h into sHA (A-swizzle scatter)
    for (int qq = 0; qq < 4; ++qq) {
      if (qq == q) continue;
      const __bf16* hxp = hx + (size_t)(((par * 2 + dir) * 2 + bg) * 4 + qq) * 2048;
      int m = tid >> 3, jl0 = (tid & 7) * 8;
      uint4 v = *(const uint4*)(hxp + m * 64 + jl0);
      const __bf16* pv = (const __bf16*)&v;
      int mt2 = m >> 4, ml2 = m & 15;
      for (int i = 0; i < 8; ++i) {
        int jg = qq * 64 + jl0 + i;
        int kt2 = jg >> 5, kk2 = jg & 31;
        int lane2 = ml2 | (((kk2 >> 3) & 1) << 4);
        int el2 = (kk2 & 7) | (((kk2 >> 4) & 1) << 3);
        sHA[(size_t)(mt2 * 8 + kt2) * 512 + lane2 * 16 + el2] = pv[i];
      }
    }
    __syncthreads();
  }
}

// ---------------- kernel 3: emissions = h @ W_cls^T + b_cls (N padded to 16) ----------------
__global__ __launch_bounds__(256) void k_emis(const float* __restrict__ bcls, unsigned char* ws) {
  __shared__ __bf16 sC[16 * 512];        // 16 KB B-swizzled classifier weights
  int tid = threadIdx.x, lane = tid & 31, wv = tid >> 5;
  {
    const uint4* src = (const uint4*)(ws + CLS_SW_OFF);
    uint4* dst = (uint4*)sC;
    for (int i = tid; i < 1024; i += 256) dst[i] = src[i];
  }
  __syncthreads();
  int m0 = blockIdx.x * 128 + wv * 16;
  int hi = lane >> 4;
  const __bf16* hall = (const __bf16*)(ws + HALL_OFF);
  const __bf16* arow = hall + (size_t)(m0 + (lane & 15)) * HIDC;
  v8f acc;
  for (int e = 0; e < 8; ++e) acc[e] = 0.f;
  for (int kt = 0; kt < 16; ++kt) {
    v16bf a = load_a_bf16_row(arow, kt, hi);
    v16bf bm = *(const v16bf*)(sC + (size_t)kt * 512 + lane * 16);
    acc = wmma_bf16(a, bm, acc);
  }
  int n = lane & 15;
  if (n < NL) {
    float bias = bcls[n];
    float* emis = (float*)(ws + EMIS_OFF);
    for (int r = 0; r < 8; ++r) {
      int m = m0 + hi * 8 + r;
      emis[(size_t)m * 16 + n] = acc[r] + bias;
    }
  }
}

// ---------------- kernel 4: CRF negative log-likelihood (one wave per sequence) ----------------
__global__ __launch_bounds__(32) void k_crf(const int* __restrict__ mask,
                                            const int* __restrict__ labels,
                                            const float* __restrict__ trans,
                                            const float* __restrict__ startv,
                                            const float* __restrict__ endv,
                                            const unsigned char* ws, float* out) {
  const float* emis = (const float*)(ws + EMIS_OFF);
  int b = blockIdx.x, j = threadIdx.x;
  bool jv = j < NL;
  __shared__ float sA[16];
  __shared__ float sR[32];
  float tcol[NL];
  for (int i = 0; i < NL; ++i) tcol[i] = jv ? trans[i * NL + j] : 0.f;
  float alpha = jv ? (startv[j] + emis[(size_t)b * NT * 16 + j]) : -1e30f;
  for (int t = 1; t < NT; ++t) {
    if (jv) sA[j] = alpha;
    __syncthreads();
    float ai[NL];
    for (int i = 0; i < NL; ++i) ai[i] = sA[i];
    float mx = -1e30f;
    for (int i = 0; i < NL; ++i) mx = fmaxf(mx, ai[i] + tcol[i]);
    float s = 0.f;
    for (int i = 0; i < NL; ++i) s += __expf(ai[i] + tcol[i] - mx);
    float em = jv ? emis[((size_t)b * NT + t) * 16 + j] : 0.f;
    float nxt = mx + __logf(s) + em;
    if (jv && mask[b * NT + t]) alpha = nxt;
    __syncthreads();
  }
  if (jv) sA[j] = alpha + endv[j];
  float p = 0.f;                          // numerator partial sums, t strided over lanes
  for (int t = 1 + j; t < NT; t += 32) {
    if (mask[b * NT + t]) {
      int yp = labels[b * NT + t - 1], yc = labels[b * NT + t];
      p += trans[yp * NL + yc] + emis[((size_t)b * NT + t) * 16 + yc];
    }
  }
  sR[j] = p;
  __syncthreads();
  if (j == 0) {
    float mx = sA[0];
    for (int i = 1; i < NL; ++i) mx = fmaxf(mx, sA[i]);
    float s = 0.f;
    for (int i = 0; i < NL; ++i) s += __expf(sA[i] - mx);
    float logZ = mx + __logf(s);
    float num = 0.f;
    for (int l2 = 0; l2 < 32; ++l2) num += sR[l2];
    int y0 = labels[b * NT];
    num += startv[y0] + emis[(size_t)b * NT * 16 + y0];
    int cm = 0;
    for (int t = 0; t < NT; ++t) cm += (mask[b * NT + t] != 0);
    int last = cm > 0 ? cm - 1 : 0;
    num += endv[labels[b * NT + last]];
    atomicAdd(out, (logZ - num) * (1.0f / NB));
  }
}

// ---------------- host launch ----------------
extern "C" void kernel_launch(void* const* d_in, const int* in_sizes, int n_in,
                              void* d_out, int out_size, void* d_ws, size_t ws_size,
                              hipStream_t stream) {
  const int*   ids    = (const int*)d_in[0];
  const int*   amask  = (const int*)d_in[1];
  const int*   labels = (const int*)d_in[2];
  const float* emb    = (const float*)d_in[3];
  const float* wihf   = (const float*)d_in[4];
  const float* whhf   = (const float*)d_in[5];
  const float* bihf   = (const float*)d_in[6];
  const float* bhhf   = (const float*)d_in[7];
  const float* wihb   = (const float*)d_in[8];
  const float* whhb   = (const float*)d_in[9];
  const float* bihb   = (const float*)d_in[10];
  const float* bhhb   = (const float*)d_in[11];
  const float* wcls   = (const float*)d_in[12];
  const float* bcls   = (const float*)d_in[13];
  const float* trans  = (const float*)d_in[14];
  const float* startv = (const float*)d_in[15];
  const float* endv   = (const float*)d_in[16];
  unsigned char* ws = (unsigned char*)d_ws;

  hipMemsetAsync(ws + SYNC_OFF, 0, 256, stream);        // spin-barrier counters
  hipMemsetAsync(d_out, 0, sizeof(float), stream);      // accumulator for mean NLL

  const int PREP_N = 2048 * 256 + 2 * 1024 * 256 + 16 * 512 + 2048;
  k_prep<<<(PREP_N + 255) / 256, 256, 0, stream>>>(wihf, whhf, bihf, bhhf,
                                                   wihb, whhb, bihb, bhhb, wcls, ws);
  k_xg<<<dim3(NT, NALL / 128), 256, 0, stream>>>(ids, emb, ws);
  k_lstm<<<16, 256, 0, stream>>>(ws);
  k_emis<<<MROWS / 128, 256, 0, stream>>>(bcls, ws);
  k_crf<<<NB, 32, 0, stream>>>(amask, labels, trans, startv, endv, ws, (float*)d_out);
}
